// TileGraphEncoder_85633057947709
// MI455X (gfx1250) — matrix operand order repaired and checked by reading, not compiled, not run on hardware
//
#include <hip/hip_runtime.h>
#include <math.h>

#define N_NODES   50000
#define N_EDGES   800000
#define EA        (N_EDGES + N_NODES)   // edges + self loops
#define IN_DIM    256
#define HID       64
#define HEADS     4
#define NUM_GRAPHS 8
#define NEG_SLOPE 0.2f
#define MT        4                     // M sub-tiles (16 rows each) per wave

typedef __attribute__((ext_vector_type(16))) __bf16 v16bf;
typedef __attribute__((ext_vector_type(8)))  float  v8f;

// ---------------------------------------------------------------------------
// WMMA GEMM: C[M,N] = A[M,K] x B[K,N]; fp32 in memory, bf16 fragments,
// fp32 accumulate via v_wmma_f32_16x16x32_bf16.
// Each wave32 computes a 64x16 block of C (4 M-tiles sharing one B fragment
// per k-step -> 4 back-to-back WMMAs, ~12 loads per WMMA instead of 32).
// Requires M%16==0, N%16==0, K%32==0 (M=50000=3125*16, K=256 here).
// Fragment layouts used here (A and B share the same contiguous K order):
//   A (16x32 bf16): half=lane>>4, row=lane&15;
//                   K = half*16 + e  (K-pairs contiguous)
//   B (32x16 bf16): col=lane&15, K = half*16 + e
//   C (16x16 f32) : col=lane&15, row(v) = v + 8*half
// ---------------------------------------------------------------------------
__global__ void gat_gemm_wmma_bf16(const float* __restrict__ A,
                                   const float* __restrict__ B,
                                   float* __restrict__ C,
                                   int M, int N, int K,
                                   int tilesN, int tilesM, int totalWaves)
{
    const int wave = (int)((blockIdx.x * blockDim.x + threadIdx.x) >> 5);
    if (wave >= totalWaves) return;           // wave-uniform -> EXEC all ones
    const int lane = threadIdx.x & 31;
    const int half = lane >> 4;
    const int r    = lane & 15;
    const int tn   = wave % tilesN;
    const int tm0  = (wave / tilesN) * MT;    // first M tile of this wave
    const int nsub = (tilesM - tm0 < MT) ? (tilesM - tm0) : MT;  // wave-uniform
    const int col  = tn * 16 + r;             // B/C column handled by this lane

    v8f acc0 = {}, acc1 = {}, acc2 = {}, acc3 = {};

    for (int k0 = 0; k0 < K; k0 += 32) {
        // ---- B fragment (shared by all 4 M sub-tiles) ----
        v16bf b;
        const float* bk = B + (size_t)(k0 + half * 16) * N + col;
#pragma unroll
        for (int e = 0; e < 16; ++e)
            b[e] = (__bf16)bk[(size_t)e * N];
        if (k0 + 32 < K)
            __builtin_prefetch(B + (size_t)(k0 + 32 + half * 16) * N + col, 0, 3);

        // ---- A fragments + WMMA per sub-tile (uniform guards keep EXEC=~0) ----
#pragma unroll
        for (int s = 0; s < MT; ++s) {
            if (s < nsub) {
                const int row = (tm0 + s) * 16 + r;
                const float* arow = A + (size_t)row * K + k0;
                v16bf a;
#pragma unroll
                for (int j = 0; j < 8; ++j) {   // K-contiguous pairs -> float2
                    const int ko = half * 16 + 2 * j;
                    const float2 f = *(const float2*)(arow + ko);
                    a[2 * j]     = (__bf16)f.x;
                    a[2 * j + 1] = (__bf16)f.y;
                }
                if (k0 + 32 < K) __builtin_prefetch(arow + 32, 0, 3);
                v8f* pacc = (s == 0) ? &acc0 : (s == 1) ? &acc1 : (s == 2) ? &acc2 : &acc3;
                *pacc = __builtin_amdgcn_wmma_f32_16x16x32_bf16(
                            false, a, false, b, (short)0, *pacc, false, false);
            }
        }
    }

#pragma unroll
    for (int s = 0; s < MT; ++s) {
        if (s < nsub) {
            const v8f accv = (s == 0) ? acc0 : (s == 1) ? acc1 : (s == 2) ? acc2 : acc3;
#pragma unroll
            for (int v = 0; v < 8; ++v) {
                const int m = (tm0 + s) * 16 + v + 8 * half;
                C[(size_t)m * N + col] = accv[v];
            }
        }
    }
}

// ---------------------------------------------------------------------------
// Elementwise fill (init accumulators / -inf for segment max)
// ---------------------------------------------------------------------------
__global__ void gat_fill(float* __restrict__ p, float v, int n)
{
    int i = blockIdx.x * blockDim.x + threadIdx.x;
    if (i < n) p[i] = v;
}

// ---------------------------------------------------------------------------
// al[n,h] = sum_c xw[n,h,c]*a_src[h,c] ; ar likewise. One thread per (n,h).
// ---------------------------------------------------------------------------
__global__ void gat_node_alpha(const float* __restrict__ xw,
                               const float* __restrict__ a_src,
                               const float* __restrict__ a_dst,
                               float* __restrict__ al,
                               float* __restrict__ ar,
                               int H)
{
    int t = blockIdx.x * blockDim.x + threadIdx.x;
    if (t >= N_NODES * H) return;
    int n = t / H, h = t % H;
    const float* xr = xw + (size_t)n * H * HID + (size_t)h * HID;
    const float* as = a_src + (size_t)h * HID;
    const float* ad = a_dst + (size_t)h * HID;
    float sl = 0.f, sr = 0.f;
#pragma unroll 8
    for (int c = 0; c < HID; ++c) { sl += xr[c] * as[c]; sr += xr[c] * ad[c]; }
    al[t] = sl;
    ar[t] = sr;
}

__device__ __forceinline__ void atomicMaxF(float* addr, float val)
{
    if (val >= 0.f)
        atomicMax((int*)addr, __float_as_int(val));
    else
        atomicMin((unsigned int*)addr, (unsigned int)__float_as_int(val));
}

__device__ __forceinline__ void edge_sd(const int* __restrict__ ei, int i,
                                        int& s, int& d)
{
    if (i < N_EDGES) { s = ei[i]; d = ei[N_EDGES + i]; }
    else             { s = d = i - N_EDGES; }            // self loop
}

// ---------------------------------------------------------------------------
// e = leaky_relu(al[src]+ar[dst]); store; segment-max into m[dst,h]
// ---------------------------------------------------------------------------
__global__ void gat_edge_logits(const int* __restrict__ ei,
                                const float* __restrict__ al,
                                const float* __restrict__ ar,
                                float* __restrict__ elog,
                                float* __restrict__ m,
                                int H)
{
    int i = blockIdx.x * blockDim.x + threadIdx.x;
    if (i >= EA) return;
    int s, d; edge_sd(ei, i, s, d);
    for (int h = 0; h < H; ++h) {
        float v = al[s * H + h] + ar[d * H + h];
        v = (v > 0.f) ? v : NEG_SLOPE * v;
        elog[(size_t)i * H + h] = v;
        atomicMaxF(&m[d * H + h], v);
    }
}

// ---------------------------------------------------------------------------
// w = exp(e - m[dst]); store over elog; segment-sum into den[dst,h]
// ---------------------------------------------------------------------------
__global__ void gat_edge_exp(const int* __restrict__ ei,
                             float* __restrict__ elog,
                             const float* __restrict__ m,
                             float* __restrict__ den,
                             int H)
{
    int i = blockIdx.x * blockDim.x + threadIdx.x;
    if (i >= EA) return;
    int s, d; edge_sd(ei, i, s, d);
    (void)s;
    for (int h = 0; h < H; ++h) {
        float w = __expf(elog[(size_t)i * H + h] - m[d * H + h]);
        elog[(size_t)i * H + h] = w;
        atomicAdd(&den[d * H + h], w);
    }
}

// ---------------------------------------------------------------------------
// One wave per edge: acc[dst, h*C+c] += (w/den[dst,h]) * xw[src, h*C+c]
// HC = H*HID (256 or 64); h = c>>6 since HID==64 in both layers.
// ---------------------------------------------------------------------------
__global__ void gat_edge_scatter(const int* __restrict__ ei,
                                 const float* __restrict__ elog,
                                 const float* __restrict__ den,
                                 const float* __restrict__ xw,
                                 float* __restrict__ acc,
                                 int H, int HC)
{
    int e = (int)((blockIdx.x * blockDim.x + threadIdx.x) >> 5);
    if (e >= EA) return;
    int lane = threadIdx.x & 31;
    int s, d; edge_sd(ei, e, s, d);
    const float* xs = xw  + (size_t)s * HC;
    float*       ad = acc + (size_t)d * HC;
    for (int c = lane; c < HC; c += 32) {
        int h = c >> 6;   // HID == 64
        float alpha = elog[(size_t)e * H + h] / den[d * H + h];
        atomicAdd(&ad[c], alpha * xs[c]);
    }
}

// ---------------------------------------------------------------------------
// out = elu(acc + bias), elementwise; may run in place (acc==out)
// ---------------------------------------------------------------------------
__global__ void gat_bias_elu(const float* __restrict__ acc,
                             const float* __restrict__ bias,
                             float* __restrict__ out, int D, int n)
{
    int i = blockIdx.x * blockDim.x + threadIdx.x;
    if (i >= n) return;
    float x = acc[i] + bias[i % D];
    out[i] = (x > 0.f) ? x : (__expf(x) - 1.f);
}

// ---------------------------------------------------------------------------
// Global mean pool accumulation: one wave per node, 64 channels
// ---------------------------------------------------------------------------
__global__ void gat_pool(const float* __restrict__ h,
                         const int* __restrict__ batch,
                         float* __restrict__ pooled,
                         float* __restrict__ cnt)
{
    int n = (int)((blockIdx.x * blockDim.x + threadIdx.x) >> 5);
    if (n >= N_NODES) return;
    int lane = threadIdx.x & 31;
    int g = batch[n];
    if (lane == 0) atomicAdd(&cnt[g], 1.f);
#pragma unroll
    for (int c = lane; c < HID; c += 32)
        atomicAdd(&pooled[g * HID + c], h[(size_t)n * HID + c]);
}

// ---------------------------------------------------------------------------
// logits[g,j] = (pooled[g]/max(cnt,1)) @ fcW + fcb  -- 16 threads total
// ---------------------------------------------------------------------------
__global__ void gat_logits(const float* __restrict__ pooled,
                           const float* __restrict__ cnt,
                           const float* __restrict__ fcW,
                           const float* __restrict__ fcb,
                           float* __restrict__ out)
{
    int t = threadIdx.x;
    if (t >= NUM_GRAPHS * 2) return;
    int g = t >> 1, j = t & 1;
    float inv = 1.f / fmaxf(cnt[g], 1.f);
    float s = fcb[j];
    for (int c = 0; c < HID; ++c)
        s += pooled[g * HID + c] * inv * fcW[c * 2 + j];
    out[g * 2 + j] = s;
}

// ---------------------------------------------------------------------------
extern "C" void kernel_launch(void* const* d_in, const int* in_sizes, int n_in,
                              void* d_out, int out_size, void* d_ws, size_t ws_size,
                              hipStream_t stream)
{
    const float* x      = (const float*)d_in[0];
    const int*   ei     = (const int*)  d_in[1];
    const int*   batch  = (const int*)  d_in[2];
    const float* W1     = (const float*)d_in[3];
    const float* a_src1 = (const float*)d_in[4];
    const float* a_dst1 = (const float*)d_in[5];
    const float* b1     = (const float*)d_in[6];
    const float* W2     = (const float*)d_in[7];
    const float* a_src2 = (const float*)d_in[8];
    const float* a_dst2 = (const float*)d_in[9];
    const float* b2     = (const float*)d_in[10];
    const float* fcW    = (const float*)d_in[11];
    const float* fcb    = (const float*)d_in[12];
    float* out = (float*)d_out;

    // ---- workspace layout (floats) ----
    float* W = (float*)d_ws;
    size_t o = 0;
    float* xw1  = W + o; o += (size_t)N_NODES * HEADS * HID;  // 12.8M
    float* acc1 = W + o; o += (size_t)N_NODES * HEADS * HID;  // 12.8M (becomes h1)
    float* al1  = W + o; o += (size_t)N_NODES * HEADS;
    float* ar1  = W + o; o += (size_t)N_NODES * HEADS;
    float* m1   = W + o; o += (size_t)N_NODES * HEADS;
    float* den1 = W + o; o += (size_t)N_NODES * HEADS;
    float* el1  = W + o; o += (size_t)EA * HEADS;             // 3.4M
    float* xw2  = W + o; o += (size_t)N_NODES * HID;          // 3.2M
    float* al2  = W + o; o += (size_t)N_NODES;
    float* ar2  = W + o; o += (size_t)N_NODES;
    float* m2   = W + o; o += (size_t)N_NODES;
    float* den2 = W + o; o += (size_t)N_NODES;
    float* el2  = W + o; o += (size_t)EA;
    float* pooled = W + o; o += NUM_GRAPHS * HID;
    float* cnt    = W + o; o += NUM_GRAPHS;
    (void)ws_size; (void)n_in; (void)in_sizes; (void)out_size;

    const int TB = 256;
    auto blocks = [](long n, int tb) { return (int)((n + tb - 1) / tb); };

    // ---- init accumulators ----
    gat_fill<<<blocks((long)N_NODES*HEADS*HID, TB), TB, 0, stream>>>(acc1, 0.f, N_NODES*HEADS*HID);
    gat_fill<<<blocks((long)N_NODES*HEADS, TB), TB, 0, stream>>>(m1, -INFINITY, N_NODES*HEADS);
    gat_fill<<<blocks((long)N_NODES*HEADS, TB), TB, 0, stream>>>(den1, 0.f, N_NODES*HEADS);
    gat_fill<<<blocks((long)N_NODES, TB), TB, 0, stream>>>(m2, -INFINITY, N_NODES);
    gat_fill<<<blocks((long)N_NODES, TB), TB, 0, stream>>>(den2, 0.f, N_NODES);
    gat_fill<<<blocks((long)N_NODES*HID, TB), TB, 0, stream>>>(out, 0.f, N_NODES*HID); // acc2
    gat_fill<<<blocks(NUM_GRAPHS*HID + NUM_GRAPHS, TB), TB, 0, stream>>>(pooled, 0.f, NUM_GRAPHS*HID + NUM_GRAPHS);

    const int tilesM = N_NODES / 16;                          // 3125
    const int groupsM = (tilesM + MT - 1) / MT;               // 782

    // ================= Layer 1: H=4, C=64 =================
    {   // xw1 = x @ W1  (M=50000, N=256, K=256) via WMMA bf16
        const int tilesN = (HEADS*HID) / 16;                  // 16
        const int total  = groupsM * tilesN;                  // 12512 waves
        gat_gemm_wmma_bf16<<<blocks((long)total*32, TB), TB, 0, stream>>>(
            x, W1, xw1, N_NODES, HEADS*HID, IN_DIM, tilesN, tilesM, total);
    }
    gat_node_alpha<<<blocks((long)N_NODES*HEADS, TB), TB, 0, stream>>>(
        xw1, a_src1, a_dst1, al1, ar1, HEADS);
    gat_edge_logits<<<blocks(EA, TB), TB, 0, stream>>>(ei, al1, ar1, el1, m1, HEADS);
    gat_edge_exp<<<blocks(EA, TB), TB, 0, stream>>>(ei, el1, m1, den1, HEADS);
    gat_edge_scatter<<<blocks((long)EA*32, TB), TB, 0, stream>>>(
        ei, el1, den1, xw1, acc1, HEADS, HEADS*HID);
    gat_bias_elu<<<blocks((long)N_NODES*HEADS*HID, TB), TB, 0, stream>>>(
        acc1, b1, acc1, HEADS*HID, N_NODES*HEADS*HID);        // acc1 now holds h1

    // ================= Layer 2: H=1, C=64 =================
    {   // xw2 = h1 @ W2  (M=50000, N=64, K=256)
        const int tilesN = HID / 16;                          // 4
        const int total  = groupsM * tilesN;                  // 3128 waves
        gat_gemm_wmma_bf16<<<blocks((long)total*32, TB), TB, 0, stream>>>(
            acc1, W2, xw2, N_NODES, HID, HEADS*HID, tilesN, tilesM, total);
    }
    gat_node_alpha<<<blocks((long)N_NODES, TB), TB, 0, stream>>>(
        xw2, a_src2, a_dst2, al2, ar2, 1);
    gat_edge_logits<<<blocks(EA, TB), TB, 0, stream>>>(ei, al2, ar2, el2, m2, 1);
    gat_edge_exp<<<blocks(EA, TB), TB, 0, stream>>>(ei, el2, m2, den2, 1);
    gat_edge_scatter<<<blocks((long)EA*32, TB), TB, 0, stream>>>(
        ei, el2, den2, xw2, out, 1, HID);                     // accumulate into d_out
    gat_bias_elu<<<blocks((long)N_NODES*HID, TB), TB, 0, stream>>>(
        out, b2, out, HID, N_NODES*HID);                      // h (first output) in place

    // ================= Pool + FC =================
    gat_pool<<<blocks((long)N_NODES*32, TB), TB, 0, stream>>>(out, batch, pooled, cnt);
    gat_logits<<<1, 32, 0, stream>>>(pooled, cnt, fcW, fcb, out + (size_t)N_NODES*HID);
}
